// MyModel_87522843559977
// MI455X (gfx1250) — compile-verified, matrix operand order checked by reading, not run on hardware
//
#include <hip/hip_runtime.h>
#include <hip/hip_bf16.h>

// ---------------------------------------------------------------------------
// GRU (B=512, T=512, F=128, UNITS=50) for gfx1250 / MI455X.
// Phase 1: time-parallel input projections Xp[t][g][b][n] via bf16 WMMA.
// Phase 2: single-WGP LDS-resident recurrent scan via bf16 WMMA + fused
//          gate math + fused final dense layer.
// ---------------------------------------------------------------------------

typedef __attribute__((ext_vector_type(16))) __bf16 v16bf;
typedef __attribute__((ext_vector_type(8)))  __bf16 v8bf;
typedef __attribute__((ext_vector_type(8)))  float  v8f;
typedef __attribute__((ext_vector_type(4)))  float  v4f;

#define B_  512
#define T_  512
#define F_  128
#define U_  50
#define NP  64      // per-gate padded hidden width (WMMA K/N padding)
#define G_  3

union FragBF { v16bf v; v8bf h[2]; __bf16 e[16]; };
union AccF   { v8f  v; float e[8]; };

static __device__ __forceinline__ unsigned short f2bf(float f) {
  __bf16 h = (__bf16)f;                       // RNE convert
  return __builtin_bit_cast(unsigned short, h);
}
static __device__ __forceinline__ float bf2f(unsigned short u) {
  unsigned int v = ((unsigned int)u) << 16;
  return __builtin_bit_cast(float, v);
}
static __device__ __forceinline__ float sigmoidf_(float x) {
  return 1.0f / (1.0f + __expf(-x));
}

// ---------------------------------------------------------------------------
// Phase 1: Xp[t][g][b][n] = bi[g*50+n] + sum_f x[b][t][f]*dp[g][b][f]*W[f][g*50+n]
// Stored bf16, n padded to 64 (pad cols are exactly 0). One wave owns one
// 16-row tile (t fixed, 16 consecutive b) and sweeps 3 gates x 4 n-tiles,
// K=128 -> 4 WMMA k-steps. W^T lives in LDS as bf16 (48 KB).
// ---------------------------------------------------------------------------
__global__ void __launch_bounds__(256)
gru_xproj_kernel(const float* __restrict__ x,     // (B,T,F)
                 const float* __restrict__ W,     // (F,150)
                 const float* __restrict__ bvec,  // (2,150): bi then br
                 const float* __restrict__ dp,    // (3,B,F)
                 unsigned short* __restrict__ Xp) // (T,3,B,NP) bf16
{
  extern __shared__ unsigned short wt[];          // [3][NP][F_] = 48 KB
  const int tid = threadIdx.x;

  for (int i = tid; i < G_ * NP * F_; i += blockDim.x) {
    const int k = i & (F_ - 1);
    const int n = (i >> 7) & (NP - 1);
    const int g = i >> 13;
    const float v = (n < U_) ? W[k * (3 * U_) + g * U_ + n] : 0.0f;
    wt[i] = f2bf(v);
  }
  __syncthreads();

  const int lane  = tid & 31;
  const int wave  = (blockIdx.x * blockDim.x + tid) >> 5;
  const int nwave = (gridDim.x * blockDim.x) >> 5;
  const int lrow  = lane & 15;   // tile row (A) / tile col (B,D)
  const int khalf = lane >> 4;   // which K half-group this lane supplies

  const int NT = (T_ * B_) / 16;            // 16384 row tiles
  for (int rt = wave; rt < NT; rt += nwave) {
    const int t  = rt >> 5;                 // B_/16 == 32 tiles per t
    const int b0 = (rt & 31) << 4;
    const int b  = b0 + lrow;
    const float* xrow = x + ((size_t)b * T_ + t) * F_;

    for (int g = 0; g < G_; ++g) {
      const float* drow = dp + ((size_t)g * B_ + b) * F_;
      FragBF a[4];
      #pragma unroll
      for (int kc = 0; kc < 4; ++kc) {
        const int f0 = kc * 32 + khalf * 8;
        v4f xa = *(const v4f*)(xrow + f0);
        v4f xb = *(const v4f*)(xrow + f0 + 4);
        v4f xc = *(const v4f*)(xrow + f0 + 16);
        v4f xd = *(const v4f*)(xrow + f0 + 20);
        v4f da = *(const v4f*)(drow + f0);
        v4f db = *(const v4f*)(drow + f0 + 4);
        v4f dc = *(const v4f*)(drow + f0 + 16);
        v4f dd = *(const v4f*)(drow + f0 + 20);
        #pragma unroll
        for (int i = 0; i < 4; ++i) {
          a[kc].e[i]      = (__bf16)(xa[i] * da[i]);
          a[kc].e[4 + i]  = (__bf16)(xb[i] * db[i]);
          a[kc].e[8 + i]  = (__bf16)(xc[i] * dc[i]);
          a[kc].e[12 + i] = (__bf16)(xd[i] * dd[i]);
        }
      }
      #pragma unroll
      for (int nj = 0; nj < 4; ++nj) {
        const int col = nj * 16 + lrow;
        const float bias = (col < U_) ? bvec[g * U_ + col] : 0.0f;
        AccF acc;
        #pragma unroll
        for (int r = 0; r < 8; ++r) acc.e[r] = bias;
        #pragma unroll
        for (int kc = 0; kc < 4; ++kc) {
          FragBF bf;
          const unsigned short* wp =
              wt + ((g * NP + col) * F_) + kc * 32 + khalf * 8;
          bf.h[0] = *(const v8bf*)(wp);
          bf.h[1] = *(const v8bf*)(wp + 16);
          acc.v = __builtin_amdgcn_wmma_f32_16x16x32_bf16(
              false, a[kc].v, false, bf.v, (short)0, acc.v, false, false);
        }
        unsigned short* op =
            Xp + (((size_t)t * G_ + g) * B_ + b0 + khalf * 8) * NP + col;
        #pragma unroll
        for (int r = 0; r < 8; ++r) op[(size_t)r * NP] = f2bf(acc.e[r]);
      }
    }
  }
}

// ---------------------------------------------------------------------------
// Phase 2: 512-step recurrent scan, one workgroup of 1024 threads (32 waves)
// on one WGP. LDS: A = bf16 h*rdp for 3 gates [3][512][64] (192KB) + U^T
// bf16 [3][64][64] (24KB) = 216KB of the 320KB WGP LDS.
// Wave w owns batch rows [16w,16w+16); per step it computes all 3 gates of
// its 4 n-tiles (24 WMMAs), applies gate math in registers (h never leaves
// VGPRs), then republishes the masked bf16 A matrices. 2 barriers/step.
// Ends with the fused dense layer h_T @ Wd + bd.
// ---------------------------------------------------------------------------
__global__ void __launch_bounds__(1024)
gru_scan_kernel(const unsigned short* __restrict__ Xp, // (T,3,B,NP) bf16
                const float* __restrict__ Umat,        // (50,150)
                const float* __restrict__ bvec,        // (2,150)
                const float* __restrict__ rdp,         // (3,B,50)
                const float* __restrict__ Wd,          // (50)
                const float* __restrict__ bd,          // (1)
                float* __restrict__ out)               // (B)
{
  extern __shared__ char smem[];
  unsigned short* Ash = (unsigned short*)smem;                     // [3][B_][NP]
  unsigned short* UTs = (unsigned short*)(smem + G_ * B_ * NP * 2);// [3][NP][NP]

  const int tid   = threadIdx.x;
  const int lane  = tid & 31;
  const int w     = tid >> 5;      // wave 0..31 -> batch block
  const int lrow  = lane & 15;
  const int khalf = lane >> 4;

  for (int i = tid; i < G_ * B_ * NP; i += 1024) Ash[i] = 0;  // h0 = 0 (+K pad)
  for (int i = tid; i < G_ * NP * NP; i += 1024) {
    const int k = i & (NP - 1);
    const int n = (i >> 6) & (NP - 1);
    const int g = i >> 12;
    const float v = (n < U_ && k < U_) ? Umat[k * (3 * U_) + g * U_ + n] : 0.0f;
    UTs[i] = f2bf(v);
  }
  __syncthreads();

  float hreg[4][8];
  #pragma unroll
  for (int nj = 0; nj < 4; ++nj)
    #pragma unroll
    for (int r = 0; r < 8; ++r) hreg[nj][r] = 0.0f;

  float brv[4][3];
  #pragma unroll
  for (int nj = 0; nj < 4; ++nj) {
    const int col = nj * 16 + lrow;
    #pragma unroll
    for (int g = 0; g < G_; ++g)
      brv[nj][g] = (col < U_) ? bvec[3 * U_ + g * U_ + col] : 0.0f;
  }

  const int brow  = w * 16 + lrow;       // A-fragment row this lane loads
  const int orow0 = w * 16 + khalf * 8;  // first output row this lane owns

  for (int t = 0; t < T_; ++t) {
    const unsigned short* xpt = Xp + (size_t)t * G_ * B_ * NP;
    #pragma unroll
    for (int nj = 0; nj < 4; ++nj) {
      const int col = nj * 16 + lrow;
      AccF acc[3];
      #pragma unroll
      for (int g = 0; g < G_; ++g)
        #pragma unroll
        for (int r = 0; r < 8; ++r) acc[g].e[r] = brv[nj][g];

      #pragma unroll
      for (int g = 0; g < G_; ++g) {
        #pragma unroll
        for (int kc = 0; kc < 2; ++kc) {
          FragBF af, bf;
          const unsigned short* ap =
              Ash + ((g * B_ + brow) * NP) + kc * 32 + khalf * 8;
          af.h[0] = *(const v8bf*)(ap);
          af.h[1] = *(const v8bf*)(ap + 16);
          const unsigned short* bp =
              UTs + ((g * NP + col) * NP) + kc * 32 + khalf * 8;
          bf.h[0] = *(const v8bf*)(bp);
          bf.h[1] = *(const v8bf*)(bp + 16);
          acc[g].v = __builtin_amdgcn_wmma_f32_16x16x32_bf16(
              false, af.v, false, bf.v, (short)0, acc[g].v, false, false);
        }
      }
      const bool valid = (col < U_);
      #pragma unroll
      for (int r = 0; r < 8; ++r) {
        const int b = orow0 + r;
        const float xz = bf2f(xpt[((size_t)0 * B_ + b) * NP + col]);
        const float xr = bf2f(xpt[((size_t)1 * B_ + b) * NP + col]);
        const float xh = bf2f(xpt[((size_t)2 * B_ + b) * NP + col]);
        const float z  = sigmoidf_(xz + acc[0].e[r]);
        const float rr = sigmoidf_(xr + acc[1].e[r]);
        const float hh = tanhf(xh + rr * acc[2].e[r]);
        const float hn = z * hreg[nj][r] + (1.0f - z) * hh;
        hreg[nj][r] = valid ? hn : 0.0f;
      }
    }
    __syncthreads();                 // everyone done reading A for step t
    if (t < T_ - 1) {                // republish masked bf16 h for step t+1
      #pragma unroll
      for (int nj = 0; nj < 4; ++nj) {
        const int col = nj * 16 + lrow;
        if (col < U_) {
          #pragma unroll
          for (int r = 0; r < 8; ++r) {
            const int b = orow0 + r;
            const float h = hreg[nj][r];
            #pragma unroll
            for (int g = 0; g < G_; ++g) {
              const float m = rdp[((size_t)g * B_ + b) * U_ + col];
              Ash[(g * B_ + b) * NP + col] = f2bf(h * m);
            }
          }
        }
      }
    }
    __syncthreads();                 // A for step t+1 visible
  }

  // Fused final dense layer: out[b] = h_T[b,:] @ Wd + bd
  float* hbuf = (float*)smem;        // [B_][NP] f32, 128KB (reuses A region)
  #pragma unroll
  for (int nj = 0; nj < 4; ++nj) {
    const int col = nj * 16 + lrow;
    if (col < U_) {
      #pragma unroll
      for (int r = 0; r < 8; ++r) hbuf[(orow0 + r) * NP + col] = hreg[nj][r];
    }
  }
  __syncthreads();
  if (tid < B_) {
    float s = bd[0];
    for (int n = 0; n < U_; ++n) s += hbuf[tid * NP + n] * Wd[n];
    out[tid] = s;
  }
}

// ---------------------------------------------------------------------------
extern "C" void kernel_launch(void* const* d_in, const int* in_sizes, int n_in,
                              void* d_out, int out_size, void* d_ws,
                              size_t ws_size, hipStream_t stream) {
  (void)in_sizes; (void)n_in; (void)out_size; (void)ws_size;
  const float* x   = (const float*)d_in[0];   // inputs (B,T,F)
  const float* W   = (const float*)d_in[1];   // (F,150)
  const float* Um  = (const float*)d_in[2];   // (50,150)
  const float* bv  = (const float*)d_in[3];   // (2,150)
  const float* Wd  = (const float*)d_in[4];   // (50,1)
  const float* bd  = (const float*)d_in[5];   // (1,)
  const float* dp  = (const float*)d_in[6];   // (3,B,F)
  const float* rdp = (const float*)d_in[7];   // (3,B,50)

  unsigned short* Xp = (unsigned short*)d_ws; // (T,3,B,NP) bf16 ~ 96 MB (L2-resident)

  const size_t smem1 = (size_t)G_ * NP * F_ * sizeof(unsigned short); // 48 KB
  gru_xproj_kernel<<<dim3(512), dim3(256), smem1, stream>>>(x, W, bv, dp, Xp);

  const size_t smem2 = (size_t)G_ * B_ * NP * sizeof(unsigned short)  // 192 KB
                     + (size_t)G_ * NP * NP * sizeof(unsigned short); //  24 KB
  gru_scan_kernel<<<dim3(1), dim3(1024), smem2, stream>>>(
      Xp, Um, bv, rdp, Wd, bd, (float*)d_out);
}
